// DDiTBlockMPS_541165879609
// MI455X (gfx1250) — compile-verified, matrix operand order checked by reading, not run on hardware
//
#include <hip/hip_runtime.h>
#include <hip/hip_bf16.h>

// ---------------------------------------------------------------------------
// DiT block (adaLN) for MI455X / gfx1250, wave32 + WMMA bf16.
// All GEMM-shaped math goes through v_wmma_f32_16x16x32_bf16 (f32 accum).
// GEMM: 32x64 tile per wave (2 A-frags x 4 B-frags = 8 WMMA / K-step).
// Unconditional speculative global_prefetch_b8 for the next K-tile keeps the
// inner loop a single straight-line block (prefetch past-end is silently
// dropped per CDNA5 ISA; no counters, no data return).
// ---------------------------------------------------------------------------

typedef __attribute__((ext_vector_type(16))) __bf16 v16bf;
typedef __attribute__((ext_vector_type(8)))  float  v8f;

#define WMMA_BF16(a, b, c) \
  __builtin_amdgcn_wmma_f32_16x16x32_bf16(false, (a), false, (b), (short)0, (c), false, false)

constexpr int Bc = 2, Sc = 2048, Hc = 768, NHc = 12, Dc = 64, Fc = 3072;
constexpr size_t MSc = (size_t)Bc * Sc;   // 4096 rows

__device__ __forceinline__ int kpat(int e, int hi) {
  // 16-bit A/B fragment K index pattern (CDNA5 ISA 7.12.2, 16x32 16-bit A)
  return (e < 8 ? e : e + 8) + (hi ? 8 : 0);
}

__device__ __forceinline__ float gelu_tanh(float x) {
  const float k0 = 0.7978845608028654f; // sqrt(2/pi)
  const float k1 = 0.044715f;
  return 0.5f * x * (1.0f + tanhf(k0 * (x + k1 * x * x * x)));
}

// -------------------- f32 -> bf16 conversion --------------------
__global__ __launch_bounds__(256) void cvt_bf16_kernel(const float* __restrict__ src,
                                                       __bf16* __restrict__ dst, int n) {
  int i = blockIdx.x * 256 + threadIdx.x;
  if (i < n) dst[i] = (__bf16)src[i];
}

// -------------------- adaLN modulation: mods = c @ ada_w.T + ada_b ----------
__global__ __launch_bounds__(256) void ada_kernel(const float* __restrict__ c,
                                                  const float* __restrict__ w,
                                                  const float* __restrict__ bias,
                                                  float* __restrict__ mods,
                                                  int H, int O) {
  int j = blockIdx.x * 256 + threadIdx.x;
  int b = blockIdx.y;
  if (j >= O) return;
  const float* cr = c + (size_t)b * H;
  const float* wr = w + (size_t)j * H;
  float acc = bias[j];
  for (int k = 0; k < H; ++k) acc += cr[k] * wr[k];
  mods[(size_t)b * O + j] = acc;
}

// -------------------- LayerNorm + adaLN scale/shift -> bf16 -----------------
__global__ __launch_bounds__(256) void ln_mod_kernel(const float* __restrict__ x,
                                                     const float* __restrict__ w,
                                                     const float* __restrict__ mods,
                                                     __bf16* __restrict__ out,
                                                     int S, int H,
                                                     int scale_off, int shift_off) {
  int row = blockIdx.x;            // over B*S
  int b = row / S;
  const float* xr = x + (size_t)row * H;
  __shared__ float red[256];
  float s = 0.f, ss = 0.f;
  for (int j = threadIdx.x; j < H; j += 256) { float v = xr[j]; s += v; ss += v * v; }
  red[threadIdx.x] = s; __syncthreads();
  for (int o = 128; o > 0; o >>= 1) { if (threadIdx.x < o) red[threadIdx.x] += red[threadIdx.x + o]; __syncthreads(); }
  float mean = red[0] / (float)H; __syncthreads();
  red[threadIdx.x] = ss; __syncthreads();
  for (int o = 128; o > 0; o >>= 1) { if (threadIdx.x < o) red[threadIdx.x] += red[threadIdx.x + o]; __syncthreads(); }
  float var = red[0] / (float)H - mean * mean;
  float rstd = rsqrtf(var + 1e-5f);
  const float* mb = mods + (size_t)b * 6 * H;
  for (int j = threadIdx.x; j < H; j += 256) {
    float xhat = (xr[j] - mean) * rstd * w[j];
    out[(size_t)row * H + j] = (__bf16)(xhat * (1.f + mb[scale_off + j]) + mb[shift_off + j]);
  }
}

// -------------------- WMMA GEMM: C[M,N] = A[M,K] @ W[N,K]^T (+bias,gelu) ----
// One wave per block; each wave computes a 32x64 tile:
//   2 A-fragments x 4 B-fragments -> 8 WMMA per 32-wide K-step.
// mode 0: f32 out; mode 1: f32 out + bias; mode 2: bias + tanh-GELU -> bf16 out
__global__ __launch_bounds__(32) void gemm_xwT_kernel(const __bf16* __restrict__ A,
                                                      const __bf16* __restrict__ W,
                                                      const float* __restrict__ bias,
                                                      float* __restrict__ Cf,
                                                      __bf16* __restrict__ Cb,
                                                      int M, int N, int K, int mode) {
  const int lane = threadIdx.x & 31;
  const int nlo = lane & 15;
  const int hi  = lane >> 4;
  const int n0   = blockIdx.x * 64;
  const int row0 = blockIdx.y * 32;

  v8f acc[2][4] = {};
  for (int k0 = 0; k0 < K; k0 += 32) {
    v16bf a[2];
#pragma unroll
    for (int ai = 0; ai < 2; ++ai) {
      const __bf16* p = A + (size_t)(row0 + 16 * ai + nlo) * K + k0;
      __builtin_prefetch(p + 32, 0, 3);   // speculative global_prefetch_b8: next K-tile
#pragma unroll
      for (int e = 0; e < 16; ++e) a[ai][e] = p[kpat(e, hi)];
    }
#pragma unroll
    for (int j = 0; j < 4; ++j) {
      v16bf bfr;
      const __bf16* wp = W + (size_t)(n0 + 16 * j + nlo) * K + k0;
      __builtin_prefetch(wp + 32, 0, 3);
#pragma unroll
      for (int e = 0; e < 16; ++e) bfr[e] = wp[kpat(e, hi)];
      acc[0][j] = WMMA_BF16(a[0], bfr, acc[0][j]);
      acc[1][j] = WMMA_BF16(a[1], bfr, acc[1][j]);
    }
  }
#pragma unroll
  for (int ai = 0; ai < 2; ++ai) {
#pragma unroll
    for (int j = 0; j < 4; ++j) {
#pragma unroll
      for (int r = 0; r < 8; ++r) {
        int rr = row0 + 16 * ai + r + 8 * hi;
        int cc = n0 + 16 * j + nlo;
        float v = acc[ai][j][r];
        if (mode >= 1) v += bias[cc];
        if (mode == 2) Cb[(size_t)rr * N + cc] = (__bf16)gelu_tanh(v);
        else           Cf[(size_t)rr * N + cc] = v;
      }
    }
  }
}

// -------------------- RoPE + split to [b,h,s,d] bf16 ------------------------
__global__ __launch_bounds__(256) void rope_kernel(const float* __restrict__ qkv,
                                                   const float* __restrict__ cos_t,
                                                   const float* __restrict__ sin_t,
                                                   __bf16* __restrict__ Q,
                                                   __bf16* __restrict__ Ko,
                                                   __bf16* __restrict__ V,
                                                   int B, int S, int nh) {
  const int d = 64;
  int idx = blockIdx.x * 256 + threadIdx.x;   // over B*S*nh*d
  int total = B * S * nh * d;
  if (idx >= total) return;
  int dim = idx & 63;
  int t = idx >> 6;
  int head = t % nh; t /= nh;
  int s = t % S; int b = t / S;
  size_t base = (size_t)(b * S + s) * (3 * nh * d);
  int col = head * d + dim;
  float q = qkv[base + col];
  float k = qkv[base + nh * d + col];
  float v = qkv[base + 2 * nh * d + col];
  float cs = cos_t[s * d + dim], sn = sin_t[s * d + dim];
  int pair = (dim < 32) ? dim + 32 : dim - 32;
  float qp = qkv[base + head * d + pair];
  float kp = qkv[base + nh * d + head * d + pair];
  float qr = (dim < 32) ? -qp : qp;
  float kr = (dim < 32) ? -kp : kp;
  size_t o = ((size_t)(b * nh + head) * S + s) * d + dim;
  Q[o]  = (__bf16)(q * cs + qr * sn);
  Ko[o] = (__bf16)(k * cs + kr * sn);
  V[o]  = (__bf16)v;
}

// -------------------- Flash attention: one wave = 16 queries ----------------
__global__ __launch_bounds__(32) void attn_kernel(const __bf16* __restrict__ Q,
                                                  const __bf16* __restrict__ K,
                                                  const __bf16* __restrict__ V,
                                                  __bf16* __restrict__ O,
                                                  int S, int nh, int Hdim) {
  const int d = 64;
  int bh = blockIdx.y;
  int q0 = blockIdx.x * 16;
  int b = bh / nh, head = bh % nh;
  const __bf16* Qp = Q + (size_t)bh * S * d;
  const __bf16* Kp = K + (size_t)bh * S * d;
  const __bf16* Vp = V + (size_t)bh * S * d;
  const int lane = threadIdx.x & 31;
  const int nlo = lane & 15;
  const int hi  = lane >> 4;

  __shared__ __align__(16) __bf16 Ks[32 * 64];
  __shared__ __align__(16) __bf16 Vs[32 * 64];
  __shared__ __align__(16) __bf16 Ps[16 * 32];

  // Q A-fragments, rows q0..q0+15, K-slices [0,32) and [32,64)
  v16bf qa0, qa1;
  {
    const __bf16* p = Qp + (size_t)(q0 + nlo) * d;
#pragma unroll
    for (int e = 0; e < 16; ++e) { int kk = kpat(e, hi); qa0[e] = p[kk]; qa1[e] = p[32 + kk]; }
  }

  v8f o[4] = {};
  float mrow[8], lrow[8];
#pragma unroll
  for (int r = 0; r < 8; ++r) { mrow[r] = -1e30f; lrow[r] = 0.f; }
  const float scale = 0.125f; // 1/sqrt(64)

  for (int kb = 0; kb < S; kb += 32) {
    // Cooperative 128-bit staged copies of K/V tiles (32 keys x 64 dims)
    const uint4* ksrc = (const uint4*)(Kp + (size_t)kb * d);
    const uint4* vsrc = (const uint4*)(Vp + (size_t)kb * d);
    uint4* kdst = (uint4*)Ks;
    uint4* vdst = (uint4*)Vs;
#pragma unroll
    for (int i = 0; i < 8; ++i) {
      kdst[lane + 32 * i] = ksrc[lane + 32 * i];
      vdst[lane + 32 * i] = vsrc[lane + 32 * i];
    }
    // speculative prefetch of next K/V tiles (past-end is silently dropped)
    __builtin_prefetch(Kp + (size_t)(kb + 32) * d + lane * 64, 0, 3);
    __builtin_prefetch(Vp + (size_t)(kb + 32) * d + lane * 64, 0, 3);
    __syncthreads();

    // scores S16x32 = Q(16x64) @ K_tile^T(64x32), two 16-col groups
    v8f sc[2] = {};
#pragma unroll
    for (int g = 0; g < 2; ++g) {
      v16bf bk0, bk1;
      const __bf16* kp_ = Ks + (size_t)(g * 16 + nlo) * d;
#pragma unroll
      for (int e = 0; e < 16; ++e) { int kk = kpat(e, hi); bk0[e] = kp_[kk]; bk1[e] = kp_[32 + kk]; }
      sc[g] = WMMA_BF16(qa0, bk0, sc[g]);
      sc[g] = WMMA_BF16(qa1, bk1, sc[g]);
    }

    // ---- online softmax (rows live across 16-lane halves) ----
    float mx[8];
#pragma unroll
    for (int r = 0; r < 8; ++r) {
      float a = sc[0][r] * scale, bv = sc[1][r] * scale;
      sc[0][r] = a; sc[1][r] = bv;
      mx[r] = fmaxf(a, bv);
    }
#pragma unroll
    for (int off = 8; off > 0; off >>= 1)
#pragma unroll
      for (int r = 0; r < 8; ++r) mx[r] = fmaxf(mx[r], __shfl_xor(mx[r], off, 32));

    float alpha[8];
#pragma unroll
    for (int r = 0; r < 8; ++r) {
      float mn = fmaxf(mrow[r], mx[r]);
      alpha[r] = __expf(mrow[r] - mn);
      mrow[r] = mn;
    }
    float rs[8];
#pragma unroll
    for (int r = 0; r < 8; ++r) rs[r] = 0.f;
#pragma unroll
    for (int g = 0; g < 2; ++g)
#pragma unroll
      for (int r = 0; r < 8; ++r) {
        float p = __expf(sc[g][r] - mrow[r]);
        Ps[(r + 8 * hi) * 32 + nlo + 16 * g] = (__bf16)p;  // -> row-major 16x32
        rs[r] += p;
      }
#pragma unroll
    for (int off = 8; off > 0; off >>= 1)
#pragma unroll
      for (int r = 0; r < 8; ++r) rs[r] += __shfl_xor(rs[r], off, 32);
#pragma unroll
    for (int r = 0; r < 8; ++r) lrow[r] = lrow[r] * alpha[r] + rs[r];
#pragma unroll
    for (int j = 0; j < 4; ++j)
#pragma unroll
      for (int r = 0; r < 8; ++r) o[j][r] *= alpha[r];
    __syncthreads();

    // P A-fragment from LDS, then O(16x64) += P(16x32) @ V(32x64)
    v16bf pa;
    {
      const __bf16* pp = Ps + (size_t)nlo * 32;
#pragma unroll
      for (int e = 0; e < 16; ++e) pa[e] = pp[kpat(e, hi)];
    }
#pragma unroll
    for (int j = 0; j < 4; ++j) {
      v16bf vb;
#pragma unroll
      for (int e = 0; e < 16; ++e) vb[e] = Vs[(size_t)kpat(e, hi) * 64 + j * 16 + nlo];
      o[j] = WMMA_BF16(pa, vb, o[j]);
    }
    __syncthreads();
  }

  // epilogue: normalize and write o as bf16 [B*S, H] at column head*64
#pragma unroll
  for (int j = 0; j < 4; ++j)
#pragma unroll
    for (int r = 0; r < 8; ++r) {
      float val = o[j][r] / lrow[r];
      size_t off = (size_t)(b * S + q0 + r + 8 * hi) * Hdim + head * 64 + j * 16 + nlo;
      O[off] = (__bf16)val;
    }
}

// -------------------- gated residual: out = x + gate * y --------------------
__global__ __launch_bounds__(256) void resid_kernel(const float* __restrict__ x,
                                                    const float* __restrict__ y,
                                                    const float* __restrict__ mods,
                                                    float* __restrict__ out,
                                                    int S, int H, int gate_off,
                                                    long total) {
  long i = (long)blockIdx.x * 256 + threadIdx.x;
  if (i >= total) return;
  int j = (int)(i % H);
  int b = (int)(i / ((long)S * H));
  out[i] = x[i] + mods[(size_t)b * 6 * H + gate_off + j] * y[i];
}

// ---------------------------------------------------------------------------
// Workspace layout (bytes)
// ---------------------------------------------------------------------------
constexpr size_t al(size_t x) { return (x + 255) & ~(size_t)255; }
constexpr size_t OFF_MODS = 0;
constexpr size_t OFF_WQKV = al(OFF_MODS + (size_t)Bc * 6 * Hc * 4);
constexpr size_t OFF_WATT = al(OFF_WQKV + (size_t)3 * Hc * Hc * 2);
constexpr size_t OFF_WM1  = al(OFF_WATT + (size_t)Hc * Hc * 2);
constexpr size_t OFF_WM2  = al(OFF_WM1  + (size_t)Fc * Hc * 2);
constexpr size_t OFF_XM   = al(OFF_WM2  + (size_t)Hc * Fc * 2);   // reused for xm2
constexpr size_t OFF_Q    = al(OFF_XM + MSc * Hc * 2);
constexpr size_t OFF_K    = al(OFF_Q  + MSc * Hc * 2);
constexpr size_t OFF_V    = al(OFF_K  + MSc * Hc * 2);
constexpr size_t OFF_O    = al(OFF_V  + MSc * Hc * 2);
constexpr size_t OFF_X1   = al(OFF_O  + MSc * Hc * 2);
constexpr size_t OFF_R    = al(OFF_X1 + MSc * Hc * 4);
// Region R (37.75 MB) sequentially reused:
constexpr size_t OFF_QKV  = OFF_R;                       // f32 [MS, 3H] (gemm1 -> rope)
constexpr size_t OFF_PROJ = OFF_R;                       // f32 [MS, H]  (gemm2 -> resid1)
constexpr size_t OFF_HDN  = OFF_R;                       // bf16 [MS, F] (mlp1 -> mlp2)
constexpr size_t OFF_Y    = al(OFF_R + MSc * Fc * 2);    // f32 [MS, H]  (mlp2 -> final)

extern "C" void kernel_launch(void* const* d_in, const int* in_sizes, int n_in,
                              void* d_out, int out_size, void* d_ws, size_t ws_size,
                              hipStream_t stream) {
  (void)in_sizes; (void)n_in; (void)out_size; (void)ws_size;
  const float* x        = (const float*)d_in[0];
  const float* cos_t    = (const float*)d_in[1];
  const float* sin_t    = (const float*)d_in[2];
  const float* c        = (const float*)d_in[3];
  const float* norm1_w  = (const float*)d_in[4];
  const float* qkv_w    = (const float*)d_in[5];
  const float* attn_w   = (const float*)d_in[6];
  const float* norm2_w  = (const float*)d_in[7];
  const float* mlp_w1   = (const float*)d_in[8];
  const float* mlp_b1   = (const float*)d_in[9];
  const float* mlp_w2   = (const float*)d_in[10];
  const float* mlp_b2   = (const float*)d_in[11];
  const float* ada_w    = (const float*)d_in[12];
  const float* ada_b    = (const float*)d_in[13];
  float* out = (float*)d_out;

  char* ws = (char*)d_ws;
  float*  mods   = (float*)(ws + OFF_MODS);
  __bf16* wqkv   = (__bf16*)(ws + OFF_WQKV);
  __bf16* watt   = (__bf16*)(ws + OFF_WATT);
  __bf16* wm1    = (__bf16*)(ws + OFF_WM1);
  __bf16* wm2    = (__bf16*)(ws + OFF_WM2);
  __bf16* xm     = (__bf16*)(ws + OFF_XM);
  __bf16* qb     = (__bf16*)(ws + OFF_Q);
  __bf16* kb     = (__bf16*)(ws + OFF_K);
  __bf16* vb     = (__bf16*)(ws + OFF_V);
  __bf16* ob     = (__bf16*)(ws + OFF_O);
  float*  x1     = (float*)(ws + OFF_X1);
  float*  qkvf   = (float*)(ws + OFF_QKV);
  float*  projf  = (float*)(ws + OFF_PROJ);
  __bf16* hdn    = (__bf16*)(ws + OFF_HDN);
  float*  yf     = (float*)(ws + OFF_Y);

  const int B = Bc, S = Sc, H = Hc, NH = NHc, F = Fc;
  const int M = B * S;          // 4096 rows

  // 1) weights -> bf16
  {
    int n;
    n = 3 * H * H; cvt_bf16_kernel<<<(n + 255) / 256, 256, 0, stream>>>(qkv_w, wqkv, n);
    n = H * H;     cvt_bf16_kernel<<<(n + 255) / 256, 256, 0, stream>>>(attn_w, watt, n);
    n = F * H;     cvt_bf16_kernel<<<(n + 255) / 256, 256, 0, stream>>>(mlp_w1, wm1, n);
    n = H * F;     cvt_bf16_kernel<<<(n + 255) / 256, 256, 0, stream>>>(mlp_w2, wm2, n);
  }

  // 2) adaLN modulation vectors
  ada_kernel<<<dim3((6 * H + 255) / 256, B), 256, 0, stream>>>(c, ada_w, ada_b, mods, H, 6 * H);

  // 3) LN1 * (1+scale_msa) + shift_msa  -> bf16 xm
  ln_mod_kernel<<<M, 256, 0, stream>>>(x, norm1_w, mods, xm, S, H, /*scale=*/H, /*shift=*/0);

  // 4) QKV = xm @ qkv_w.T   (WMMA, 32x64 per wave)
  gemm_xwT_kernel<<<dim3(3 * H / 64, M / 32), 32, 0, stream>>>(
      xm, wqkv, nullptr, qkvf, nullptr, M, 3 * H, H, 0);

  // 5) RoPE + split into q/k/v (bf16, [b,h,s,d])
  {
    int total = B * S * NH * 64;
    rope_kernel<<<(total + 255) / 256, 256, 0, stream>>>(qkvf, cos_t, sin_t, qb, kb, vb, B, S, NH);
  }

  // 6) flash attention (WMMA scores + PV), output bf16 [B*S, H]
  attn_kernel<<<dim3(S / 16, B * NH), 32, 0, stream>>>(qb, kb, vb, ob, S, NH, H);

  // 7) proj = o @ attn_out_w.T   (WMMA)
  gemm_xwT_kernel<<<dim3(H / 64, M / 32), 32, 0, stream>>>(
      ob, watt, nullptr, projf, nullptr, M, H, H, 0);

  // 8) x1 = x + gate_msa * proj
  {
    long total = (long)M * H;
    resid_kernel<<<(unsigned)((total + 255) / 256), 256, 0, stream>>>(
        x, projf, mods, x1, S, H, /*gate_msa=*/2 * H, total);
  }

  // 9) LN2 * (1+scale_mlp) + shift_mlp -> bf16 xm (reused)
  ln_mod_kernel<<<M, 256, 0, stream>>>(x1, norm2_w, mods, xm, S, H, /*scale=*/4 * H, /*shift=*/3 * H);

  // 10) hdn = gelu(xm @ mlp_w1.T + b1) -> bf16  (WMMA, fused epilogue)
  gemm_xwT_kernel<<<dim3(F / 64, M / 32), 32, 0, stream>>>(
      xm, wm1, mlp_b1, nullptr, hdn, M, F, H, 2);

  // 11) y = hdn @ mlp_w2.T + b2  (WMMA)
  gemm_xwT_kernel<<<dim3(H / 64, M / 32), 32, 0, stream>>>(
      hdn, wm2, mlp_b2, yf, nullptr, M, H, F, 1);

  // 12) out = x1 + gate_mlp * y
  {
    long total = (long)M * H;
    resid_kernel<<<(unsigned)((total + 255) / 256), 256, 0, stream>>>(
        x1, yf, mods, out, S, H, /*gate_mlp=*/5 * H, total);
  }
}
